// SelfAttention_1271310320265
// MI455X (gfx1250) — matrix operand order repaired and checked by reading, compiled, not run on hardware
//
#include <hip/hip_runtime.h>
#include <hip/hip_bf16.h>

// ---------------------------------------------------------------------------
// GLM-style self-attention block on gfx1250 (CDNA5, wave32, WMMA).
//   S=2048, H=4096, HEADS=32, HEAD_DIM=128, ROT_DIM=64
// Pipeline:
//   K1 gemm:  qkv(bf16)[S,12288] = hs(f32) @ Wqkv(f32) + bqkv   (WMMA bf16)
//   K2 rope:  in-place 2D rotary on q,k halves of qkv buffer
//   K3 attn:  flash attention per (head, 64-query block)        (WMMA bf16)
//             K tile staged with GLOBAL_LOAD_ASYNC_TO_LDS when available
//   K4 gemm:  out(f32)[S,4096]  = ctx(f32) @ Wd(f32) + bd       (WMMA bf16)
// ---------------------------------------------------------------------------

typedef __attribute__((ext_vector_type(16))) __bf16 bf16x16;
typedef __attribute__((ext_vector_type(8)))  float  f32x8;
typedef int v4i __attribute__((vector_size(16)));

union BF16Frag {
    unsigned int   u32[8];
    unsigned short u16[16];
    bf16x16        v;
};

#if defined(__has_builtin)
#if __has_builtin(__builtin_amdgcn_global_load_async_to_lds_b128)
#define USE_ASYNC_LDS 1
#endif
#endif
#ifndef USE_ASYNC_LDS
#define USE_ASYNC_LDS 0
#endif

#define GLOBAL_AS __attribute__((address_space(1)))
#define LDS_AS    __attribute__((address_space(3)))

#if USE_ASYNC_LDS
__device__ __forceinline__ void async_b128(const void* g, void* l) {
    // (global src, lds dst, imm offset, imm cpol); params are v4i pointers.
    __builtin_amdgcn_global_load_async_to_lds_b128(
        (GLOBAL_AS v4i*)g, (LDS_AS v4i*)l, 0, 0);
}
__device__ __forceinline__ void wait_async0() {
#if __has_builtin(__builtin_amdgcn_s_wait_asynccnt)
    __builtin_amdgcn_s_wait_asynccnt(0);
#else
    asm volatile("s_wait_asynccnt 0" ::: "memory");
#endif
}
#endif

__device__ __forceinline__ unsigned short f32_to_bf16(float x) {
    unsigned int u = __float_as_uint(x);
    u += 0x7FFFu + ((u >> 16) & 1u);          // round-to-nearest-even
    return (unsigned short)(u >> 16);
}
__device__ __forceinline__ unsigned int pack_bf16x2(float lo, float hi) {
    return (unsigned int)f32_to_bf16(lo) | ((unsigned int)f32_to_bf16(hi) << 16);
}
__device__ __forceinline__ float bf16_to_f32(unsigned short h) {
    return __uint_as_float(((unsigned int)h) << 16);
}

// A-matrix (16x32, 16-bit) K index for fragment element e, lane half hi.
// lanes 0-15: e0..7 -> K0..7, e8..15 -> K16..23 ; lanes 16-31: +8
__device__ __forceinline__ int frag_k_A(int e, int hi) {
    return (e < 8 ? e : e + 8) + (hi << 3);
}
// B-matrix (32x16, 16-bit): same K order as A: lanes 0-15: e0..7 -> K0..7, e8..15 -> K16..23 ; lanes 16-31: +8
__device__ __forceinline__ int frag_k_B(int e, int hi) {
    return (e < 8 ? e : e + 8) + (hi << 3);
}

__device__ __forceinline__ f32x8 wmma_bf16(const BF16Frag& a, const BF16Frag& b, f32x8 c) {
    return __builtin_amdgcn_wmma_f32_16x16x32_bf16(
        false, a.v, false, b.v, (short)0, c, false, false);
}

// ---------------------------------------------------------------------------
// Tiled GEMM: C[M,N] = A[M,K](f32) * B[K,N](f32) + bias[N]
// BM=128, BN=128, BK=32; 256 threads = 8 waves (4x2), wave tile 32x64.
// ---------------------------------------------------------------------------
template <bool OUT_BF16>
__global__ __launch_bounds__(256)
void gemm_bf16_kernel(const float* __restrict__ A, const float* __restrict__ B,
                      const float* __restrict__ bias, void* __restrict__ Cout,
                      int M, int N, int K) {
    __shared__ unsigned short aLds[128][40];   // [m][k]  (row = 80B)
    __shared__ unsigned short bLds[128][40];   // [n][k]  transposed

    const int tid  = threadIdx.x;
    const int lane = tid & 31;
    const int wave = tid >> 5;
    const int lm   = lane & 15;
    const int hi   = lane >> 4;
    const int wm   = wave >> 1;         // 0..3  (row block of 32)
    const int wn   = wave & 1;          // 0..1  (col block of 64)

    const int n0 = blockIdx.x * 128;
    const int m0 = blockIdx.y * 128;

    f32x8 acc[2][4];
    #pragma unroll
    for (int mt = 0; mt < 2; ++mt)
        #pragma unroll
        for (int nt = 0; nt < 4; ++nt)
            acc[mt][nt] = {};

    for (int k0 = 0; k0 < K; k0 += 32) {
        __syncthreads();
        // A tile: 128 rows x 32 k ; thread -> (row = tid>>1, 16 cols) via float4
        {
            const int r  = tid >> 1;
            const int c0 = (tid & 1) * 16;
            const float4* src =
                (const float4*)(A + (size_t)(m0 + r) * K + k0 + c0);
            unsigned int* dst = (unsigned int*)&aLds[r][c0];
            #pragma unroll
            for (int i = 0; i < 4; ++i) {
                const float4 f = src[i];
                dst[2 * i]     = pack_bf16x2(f.x, f.y);
                dst[2 * i + 1] = pack_bf16x2(f.z, f.w);
            }
        }
        // B tile: 32 k x 128 n ; thread -> (kr = tid>>3, 16 cols), store transposed
        {
            const int kr = tid >> 3;
            const int c0 = (tid & 7) * 16;
            const float4* src =
                (const float4*)(B + (size_t)(k0 + kr) * N + n0 + c0);
            #pragma unroll
            for (int i = 0; i < 4; ++i) {
                const float4 f = src[i];
                bLds[c0 + 4 * i + 0][kr] = f32_to_bf16(f.x);
                bLds[c0 + 4 * i + 1][kr] = f32_to_bf16(f.y);
                bLds[c0 + 4 * i + 2][kr] = f32_to_bf16(f.z);
                bLds[c0 + 4 * i + 3][kr] = f32_to_bf16(f.w);
            }
        }
        // Prefetch next k-step's lines (emits global_prefetch_b8).
        if (k0 + 32 < K) {
            __builtin_prefetch(
                A + (size_t)(m0 + (tid >> 1)) * K + k0 + 32 + (tid & 1) * 16, 0, 1);
            __builtin_prefetch(
                B + (size_t)(k0 + 32 + (tid >> 3)) * N + n0 + (tid & 7) * 16, 0, 1);
        }
        __syncthreads();

        BF16Frag af[2], bf[4];
        #pragma unroll
        for (int mt = 0; mt < 2; ++mt) {
            const int row = wm * 32 + mt * 16 + lm;
            #pragma unroll
            for (int ep = 0; ep < 8; ++ep)
                af[mt].u32[ep] = *(const unsigned int*)&aLds[row][frag_k_A(2 * ep, hi)];
        }
        #pragma unroll
        for (int nt = 0; nt < 4; ++nt) {
            const int row = wn * 64 + nt * 16 + lm;
            #pragma unroll
            for (int ep = 0; ep < 8; ++ep)
                bf[nt].u32[ep] = *(const unsigned int*)&bLds[row][frag_k_B(2 * ep, hi)];
        }
        #pragma unroll
        for (int mt = 0; mt < 2; ++mt)
            #pragma unroll
            for (int nt = 0; nt < 4; ++nt)
                acc[mt][nt] = wmma_bf16(af[mt], bf[nt], acc[mt][nt]);
    }

    // Epilogue: bias + store. C/D layout: lane -> N = lm, VGPR r -> M = r + 8*hi
    #pragma unroll
    for (int nt = 0; nt < 4; ++nt) {
        const int n = n0 + wn * 64 + nt * 16 + lm;
        const float bv = bias[n];
        #pragma unroll
        for (int mt = 0; mt < 2; ++mt) {
            #pragma unroll
            for (int r = 0; r < 8; ++r) {
                const int m = m0 + wm * 32 + mt * 16 + r + 8 * hi;
                const float val = acc[mt][nt][r] + bv;
                if (OUT_BF16)
                    ((unsigned short*)Cout)[(size_t)m * N + n] = f32_to_bf16(val);
                else
                    ((float*)Cout)[(size_t)m * N + n] = val;
            }
        }
    }
}

// ---------------------------------------------------------------------------
// RoPE (2D rotary), in place on bf16 qkv buffer [S][HEADS*384].
// One thread handles one (s, head, half, d<32) pair for both q and k.
// ---------------------------------------------------------------------------
__global__ __launch_bounds__(256)
void rope_kernel(unsigned short* __restrict__ qkv, const int* __restrict__ positions) {
    const int idx  = blockIdx.x * 256 + threadIdx.x;   // [0, 2048*32*64)
    const int d    = idx & 31;
    const int half = (idx >> 5) & 1;
    const int h    = (idx >> 6) & 31;
    const int s    = idx >> 11;

    const int pos = positions[half * 2048 + s];
    // inv_freq = 10000^(-d/32) = exp(-d * ln(10000)/32)
    const float freq = __expf(-(float)d * 0.28782313662425576f);
    const float ang  = (float)pos * freq;
    float sn, cs;
    __sincosf(ang, &sn, &cs);

    const size_t base = (size_t)s * 12288 + (size_t)h * 384 + half * 64 + d;

    // q at +0, k at +128
    #pragma unroll
    for (int qk = 0; qk < 2; ++qk) {
        const size_t o = base + qk * 128;
        const float x1 = bf16_to_f32(qkv[o]);
        const float x2 = bf16_to_f32(qkv[o + 32]);
        qkv[o]      = f32_to_bf16(x1 * cs - x2 * sn);
        qkv[o + 32] = f32_to_bf16(x2 * cs + x1 * sn);
    }
}

// ---------------------------------------------------------------------------
// Flash attention: block = (head, 64-query tile), 4 waves, each wave owns
// 16 queries. kv processed in chunks of 32, K/V staged in LDS.
// ---------------------------------------------------------------------------
__global__ __launch_bounds__(128)
void attn_kernel(const unsigned short* __restrict__ qkv, float* __restrict__ ctxOut) {
    __shared__ unsigned short kLds[32][136];      // [kv][d]  (row = 272B, 16B aligned)
    __shared__ unsigned short vLds[128][34];      // [d][kv]  transposed (pad 2)
    __shared__ unsigned short pLds[4][16][34];    // per-wave P  [m][kv] (pad 2)

    const int tid   = threadIdx.x;
    const int lane  = tid & 31;
    const int wave  = tid >> 5;
    const int lm    = lane & 15;
    const int hi    = lane >> 4;
    const int head  = blockIdx.x;
    const int qbase = blockIdx.y * 64;
    const float scale = 0.08838834764831845f;     // 128^-0.5

    // Load Q fragments (A-operand), 4 k-chunks of 32 over head_dim=128.
    BF16Frag qf[4];
    {
        const int qrow = qbase + wave * 16 + lm;
        const unsigned short* gq = qkv + (size_t)qrow * 12288 + (size_t)head * 384;
        #pragma unroll
        for (int kc = 0; kc < 4; ++kc)
            #pragma unroll
            for (int ep = 0; ep < 8; ++ep)
                qf[kc].u32[ep] =
                    *(const unsigned int*)(gq + kc * 32 + frag_k_A(2 * ep, hi));
    }

    float mrow[8], lrow[8];
    f32x8 cacc[8];
    #pragma unroll
    for (int r = 0; r < 8; ++r) { mrow[r] = -1e30f; lrow[r] = 0.0f; }
    #pragma unroll
    for (int dt = 0; dt < 8; ++dt) cacc[dt] = {};

    const int kv_end = qbase + 64;
    for (int kv0 = 0; kv0 < kv_end; kv0 += 32) {
        __syncthreads();
        // Stage K chunk [32][128] (async DMA when available) and V chunk
        // transposed [128][32] into LDS.
        {
            const int r  = tid >> 2;
            const int c0 = (tid & 3) * 32;
            const unsigned short* srcK =
                qkv + (size_t)(kv0 + r) * 12288 + (size_t)head * 384 + 128 + c0;
#if USE_ASYNC_LDS
            #pragma unroll
            for (int i = 0; i < 4; ++i)
                async_b128(srcK + 8 * i, &kLds[r][c0 + 8 * i]);
#else
            unsigned int* dK = (unsigned int*)&kLds[r][c0];
            #pragma unroll
            for (int i = 0; i < 16; ++i)
                dK[i] = *(const unsigned int*)(srcK + 2 * i);
#endif
            const unsigned short* srcV =
                qkv + (size_t)(kv0 + r) * 12288 + (size_t)head * 384 + 256 + c0;
            #pragma unroll
            for (int i = 0; i < 32; ++i)
                vLds[c0 + i][r] = srcV[i];
#if USE_ASYNC_LDS
            wait_async0();
#endif
        }
        __syncthreads();

        // Scores: two 16-kv tiles, each 4 WMMAs over d.
        f32x8 sacc[2];
        sacc[0] = {}; sacc[1] = {};
        #pragma unroll
        for (int t = 0; t < 2; ++t) {
            #pragma unroll
            for (int kc = 0; kc < 4; ++kc) {
                BF16Frag kb;
                const int krow = t * 16 + lm;              // kv position (N)
                #pragma unroll
                for (int ep = 0; ep < 8; ++ep)
                    kb.u32[ep] = *(const unsigned int*)
                        &kLds[krow][kc * 32 + frag_k_B(2 * ep, hi)];
                sacc[t] = wmma_bf16(qf[kc], kb, sacc[t]);
            }
        }

        // Online softmax per row (row M = r + 8*hi spans 16 lanes of one half).
        float alphar[8];
        #pragma unroll
        for (int r = 0; r < 8; ++r) {
            const int qg = qbase + wave * 16 + r + 8 * hi;
            float s0 = sacc[0][r] * scale;
            float s1 = sacc[1][r] * scale;
            if (kv0 + lm      > qg) s0 = -1e30f;
            if (kv0 + 16 + lm > qg) s1 = -1e30f;
            float mx = fmaxf(s0, s1);
            #pragma unroll
            for (int msk = 1; msk <= 8; msk <<= 1)
                mx = fmaxf(mx, __shfl_xor(mx, msk, 32));
            const float mn    = fmaxf(mrow[r], mx);
            const float alpha = __expf(mrow[r] - mn);
            const float p0 = __expf(s0 - mn);
            const float p1 = __expf(s1 - mn);
            float ps = p0 + p1;
            #pragma unroll
            for (int msk = 1; msk <= 8; msk <<= 1)
                ps += __shfl_xor(ps, msk, 32);
            lrow[r] = lrow[r] * alpha + ps;
            mrow[r] = mn;
            alphar[r] = alpha;
            // stash P (bf16) at [m = r+8*hi][kv col]
            pLds[wave][r + 8 * hi][lm]      = f32_to_bf16(p0);
            pLds[wave][r + 8 * hi][16 + lm] = f32_to_bf16(p1);
        }

        // Rescale ctx accumulators, then PV WMMAs.
        #pragma unroll
        for (int dt = 0; dt < 8; ++dt)
            #pragma unroll
            for (int r = 0; r < 8; ++r)
                cacc[dt][r] *= alphar[r];

        BF16Frag pf;     // A-operand: P tile 16(q) x 32(kv), same wave's LDS
        #pragma unroll
        for (int ep = 0; ep < 8; ++ep)
            pf.u32[ep] = *(const unsigned int*)&pLds[wave][lm][frag_k_A(2 * ep, hi)];

        #pragma unroll
        for (int dt = 0; dt < 8; ++dt) {
            BF16Frag vb; // B-operand: V 32(kv) x 16(d); vLds is [d][kv]
            #pragma unroll
            for (int ep = 0; ep < 8; ++ep)
                vb.u32[ep] = *(const unsigned int*)
                    &vLds[dt * 16 + lm][frag_k_B(2 * ep, hi)];
            cacc[dt] = wmma_bf16(pf, vb, cacc[dt]);
        }
    }

    // Normalize and store ctx f32 [S][H] with H index = head*128 + d.
    #pragma unroll
    for (int dt = 0; dt < 8; ++dt) {
        #pragma unroll
        for (int r = 0; r < 8; ++r) {
            const int m = qbase + wave * 16 + r + 8 * hi;
            const int d = dt * 16 + lm;
            ctxOut[(size_t)m * 4096 + head * 128 + d] = cacc[dt][r] / lrow[r];
        }
    }
}

// ---------------------------------------------------------------------------
// Host-side launch
// ---------------------------------------------------------------------------
extern "C" void kernel_launch(void* const* d_in, const int* in_sizes, int n_in,
                              void* d_out, int out_size, void* d_ws, size_t ws_size,
                              hipStream_t stream) {
    (void)in_sizes; (void)n_in; (void)out_size; (void)ws_size;
    const float* hs        = (const float*)d_in[0];
    const int*   positions = (const int*)d_in[1];
    const float* Wqkv      = (const float*)d_in[2];
    const float* bqkv      = (const float*)d_in[3];
    const float* Wd        = (const float*)d_in[4];
    const float* bd        = (const float*)d_in[5];
    float*       out       = (float*)d_out;

    const int S = 2048, H = 4096, QKV = 12288;

    unsigned short* qkv = (unsigned short*)d_ws;                       // bf16 [S][12288]
    float* ctx = (float*)((char*)d_ws + (size_t)S * QKV * sizeof(unsigned short));

    // K1: qkv = hs @ Wqkv + bqkv   (bf16 out)
    gemm_bf16_kernel<true><<<dim3(QKV / 128, S / 128), 256, 0, stream>>>(
        hs, Wqkv, bqkv, qkv, S, QKV, H);

    // K2: RoPE in place on q,k halves
    rope_kernel<<<(S * 32 * 64) / 256, 256, 0, stream>>>(qkv, positions);

    // K3: flash attention -> ctx f32 [S][H]
    attn_kernel<<<dim3(32, S / 64), 128, 0, stream>>>(qkv, ctx);

    // K4: out = ctx @ Wd + bd   (f32 out)
    gemm_bf16_kernel<false><<<dim3(H / 128, S / 128), 256, 0, stream>>>(
        ctx, Wd, bd, out, S, H, H);
}
